// GNN_53085795779195
// MI455X (gfx1250) — matrix operand order repaired and checked by reading, not compiled, not run on hardware
//
#include <hip/hip_runtime.h>

typedef __attribute__((ext_vector_type(16))) _Float16 v16h;
typedef __attribute__((ext_vector_type(8)))  float    v8f;

#define D 64

__device__ __forceinline__ void atomAddF(float* p, float v) {
  // lowers to global_atomic_add_f32 (relaxed, device scope)
  (void)__hip_atomic_fetch_add(p, v, __ATOMIC_RELAXED, __HIP_MEMORY_SCOPE_AGENT);
}

// ---------------- degree / normalization ----------------
__global__ void k_init_deg(float* __restrict__ deg, int n) {
  int i = blockIdx.x * blockDim.x + threadIdx.x;
  if (i < n) deg[i] = 1.0f;  // self loop
}

__global__ void k_count_deg(float* __restrict__ deg, const int* __restrict__ col, int E) {
  int e = blockIdx.x * blockDim.x + threadIdx.x;
  if (e < E) atomAddF(&deg[col[e]], 1.0f);
}

__global__ void k_dinv(float* __restrict__ deg, int n) {
  int i = blockIdx.x * blockDim.x + threadIdx.x;
  if (i < n) deg[i] = rsqrtf(fmaxf(deg[i], 1.0f));
}

__global__ void k_edge_norm(float* __restrict__ nrm, const float* __restrict__ dinv,
                            const int* __restrict__ row, const int* __restrict__ col, int E) {
  int e = blockIdx.x * blockDim.x + threadIdx.x;
  if (e < E) nrm[e] = dinv[row[e]] * dinv[col[e]];
}

// ---------------- W pre-swizzle: f32 [3][64][64] -> f16 WMMA B fragments ----------------
// Bswz layout: [L][kk2][jt][lane][16 halfs]; each lane's fragment is one contiguous 32B chunk.
__global__ void k_prep_w(_Float16* __restrict__ Bswz, const float* __restrict__ Ws) {
  int t = blockIdx.x * blockDim.x + threadIdx.x;
  if (t >= 3 * 2 * 4 * 32) return;
  int lane = t & 31;
  int jt   = (t >> 5) & 3;
  int kk2  = (t >> 7) & 1;
  int L    = t >> 8;
  int hi  = lane >> 4;
  int N   = (lane & 15) + jt * 16;
  int kb  = kk2 * 32 + 8 * hi;
  const float* W = Ws + (size_t)L * D * D;
  v16h b;
#pragma unroll
  for (int j = 0; j < 8; ++j) {
    b[j]     = (_Float16)W[(kb + j) * D + N];
    b[8 + j] = (_Float16)W[(kb + 16 + j) * D + N];
  }
  ((v16h*)Bswz)[t] = b;
}

// ---------------- GEMM: T = H @ W (WMMA), fused agg = T * dinv^2 (self-loop init) ----------
// One wave per 16-row x 64-col tile: 4 accumulators, K=64 in two 32-steps.
__global__ void k_gemm_wmma(const float* __restrict__ H, const _Float16* __restrict__ BswzL,
                            float* __restrict__ T, float* __restrict__ agg,
                            const float* __restrict__ dinv, int n) {
  int wave = (blockIdx.x * blockDim.x + threadIdx.x) >> 5;
  int lane = threadIdx.x & 31;
  int m0 = wave << 4;
  if (m0 >= n) return;  // wave-uniform guard: EXEC stays all-ones for WMMA
  const int hi  = lane >> 4;
  const int l15 = lane & 15;
  const v16h* Bf = (const v16h*)BswzL;

  v8f acc[4] = {};
#pragma unroll
  for (int kk2 = 0; kk2 < 2; ++kk2) {
    // A fragment (16x32 f16): two contiguous 8-float runs -> 4x b128 loads, then pack cvt.
    const float4* p0 = (const float4*)(H + (size_t)(m0 + l15) * D + kk2 * 32 + 8 * hi);
    const float4* p1 = (const float4*)(H + (size_t)(m0 + l15) * D + kk2 * 32 + 16 + 8 * hi);
    float4 c0 = p0[0], c1 = p0[1], c2 = p1[0], c3 = p1[1];
    v16h a;
    a[0]  = (_Float16)c0.x; a[1]  = (_Float16)c0.y; a[2]  = (_Float16)c0.z; a[3]  = (_Float16)c0.w;
    a[4]  = (_Float16)c1.x; a[5]  = (_Float16)c1.y; a[6]  = (_Float16)c1.z; a[7]  = (_Float16)c1.w;
    a[8]  = (_Float16)c2.x; a[9]  = (_Float16)c2.y; a[10] = (_Float16)c2.z; a[11] = (_Float16)c2.w;
    a[12] = (_Float16)c3.x; a[13] = (_Float16)c3.y; a[14] = (_Float16)c3.z; a[15] = (_Float16)c3.w;
#pragma unroll
    for (int jt = 0; jt < 4; ++jt) {
      v16h b = Bf[((size_t)kk2 * 4 + jt) * 32 + lane];  // coalesced 32B/lane
      acc[jt] = __builtin_amdgcn_wmma_f32_16x16x32_f16(
          /*neg_a=*/false, a, /*neg_b=*/false, b,
          /*c_mod=*/(short)0, acc[jt], /*reuse_a=*/false, /*reuse_b=*/false);
    }
  }
  // self-loop weights for the 8 rows this lane-half owns
  float d2[8];
#pragma unroll
  for (int r = 0; r < 8; ++r) {
    float di = dinv[m0 + 8 * hi + r];
    d2[r] = di * di;
  }
  // C/D layout: VGPR r -> row m0 + r + 8*hi, col = jt*16 + l15
#pragma unroll
  for (int jt = 0; jt < 4; ++jt)
#pragma unroll
    for (int r = 0; r < 8; ++r) {
      size_t off = (size_t)(m0 + r + 8 * hi) * D + jt * 16 + l15;
      float v = acc[jt][r];
      T[off]   = v;
      agg[off] = v * d2[r];
    }
}

// ---------------- edge scatter: one wave per edge, 2 feats/lane, hw f32 atomics -----------
__global__ void k_scatter(float* __restrict__ agg, const float* __restrict__ T,
                          const float* __restrict__ nrm,
                          const int* __restrict__ row, const int* __restrict__ col, int E) {
  int t = blockIdx.x * blockDim.x + threadIdx.x;
  int e = t >> 5;
  if (e >= E) return;
  int lane = t & 31;
  int r = row[e], c = col[e];
  float w = nrm[e];
  float2 v = ((const float2*)(T + (size_t)r * D))[lane];
  float* dst = agg + (size_t)c * D + 2 * lane;
  atomAddF(dst,     v.x * w);
  atomAddF(dst + 1, v.y * w);
}

// ---------------- bias + BN + ReLU, float4 vectorized, in place ----------------
__global__ void k_bn_relu(float* __restrict__ h, const float* __restrict__ b,
                          const float* __restrict__ gamma, const float* __restrict__ beta,
                          const float* __restrict__ mean, const float* __restrict__ var, int n) {
  int t = blockIdx.x * blockDim.x + threadIdx.x;
  if (t >= n * (D / 4)) return;
  int g = t & 15;  // feature group of 4
  float4 v  = ((const float4*)h)[t];
  float4 bb = ((const float4*)b)[g];
  float4 gg = ((const float4*)gamma)[g];
  float4 be = ((const float4*)beta)[g];
  float4 mm = ((const float4*)mean)[g];
  float4 vv = ((const float4*)var)[g];
  v.x = fmaxf((v.x + bb.x - mm.x) * rsqrtf(vv.x + 1e-5f) * gg.x + be.x, 0.0f);
  v.y = fmaxf((v.y + bb.y - mm.y) * rsqrtf(vv.y + 1e-5f) * gg.y + be.y, 0.0f);
  v.z = fmaxf((v.z + bb.z - mm.z) * rsqrtf(vv.z + 1e-5f) * gg.z + be.z, 0.0f);
  v.w = fmaxf((v.w + bb.w - mm.w) * rsqrtf(vv.w + 1e-5f) * gg.w + be.w, 0.0f);
  ((float4*)h)[t] = v;
}

// ---------------- final head: out[i] = dot(h[i], w) + bias (wave32 per node) --------------
__global__ void k_final(float* __restrict__ out, const float* __restrict__ h,
                        const float* __restrict__ w, const float* __restrict__ bias, int n) {
  int t = blockIdx.x * blockDim.x + threadIdx.x;
  int i = t >> 5;
  if (i >= n) return;
  int lane = t & 31;
  float2 hv = ((const float2*)(h + (size_t)i * D))[lane];
  float2 wv = ((const float2*)w)[lane];
  float s = hv.x * wv.x + hv.y * wv.y;
#pragma unroll
  for (int m = 16; m >= 1; m >>= 1) s += __shfl_xor(s, m, 32);
  if (lane == 0) out[i] = s + bias[0];
}

extern "C" void kernel_launch(void* const* d_in, const int* in_sizes, int n_in,
                              void* d_out, int out_size, void* d_ws, size_t ws_size,
                              hipStream_t stream) {
  const float* x      = (const float*)d_in[0];
  const int*   ei     = (const int*)d_in[1];
  const float* Ws     = (const float*)d_in[2];
  const float* bs     = (const float*)d_in[3];
  const float* gammas = (const float*)d_in[4];
  const float* betas  = (const float*)d_in[5];
  const float* means  = (const float*)d_in[6];
  const float* vars   = (const float*)d_in[7];
  const float* lin_w  = (const float*)d_in[8];
  const float* lin_b  = (const float*)d_in[9];
  float* out = (float*)d_out;

  const int n = in_sizes[0] / D;   // 50000
  const int E = in_sizes[1] / 2;   // 800000
  const int* rowp = ei;            // edge_index[0] = source
  const int* colp = ei + E;        // edge_index[1] = target

  // workspace partition (~42 MB): dinv | edge norm | T | A | B | Bswz(f16)
  float* ws   = (float*)d_ws;
  float* deg  = ws;                            // n floats; becomes dinv after k_dinv
  float* nrm  = deg + ((n + 63) & ~63);        // E floats
  float* bufT = nrm + ((E + 63) & ~63);        // n*64
  float* bufA = bufT + (size_t)n * D;          // n*64
  float* bufB = bufA + (size_t)n * D;          // n*64
  _Float16* Bswz = (_Float16*)(bufB + (size_t)n * D);  // 3*2*4*32*16 halfs = 24 KB

  const int B = 256;  // 8 waves/block on wave32
  auto blocks = [&](long long work) { return (int)((work + B - 1) / B); };

  // ---- graph normalization + weight swizzle (once, amortized over all layers) ----
  k_init_deg <<<blocks(n), B, 0, stream>>>(deg, n);
  k_count_deg<<<blocks(E), B, 0, stream>>>(deg, colp, E);
  k_dinv     <<<blocks(n), B, 0, stream>>>(deg, n);
  k_edge_norm<<<blocks(E), B, 0, stream>>>(nrm, deg, rowp, colp, E);
  k_prep_w   <<<blocks(3 * 2 * 4 * 32), B, 0, stream>>>(Bswz, Ws);

  // ---- 3 GCN layers ----
  const float* hin = x;
  float* hbuf[2] = {bufA, bufB};
  for (int L = 0; L < 3; ++L) {
    float* agg = hbuf[L & 1];
    long long gemmThreads = (long long)((n + 15) / 16) * 32;  // one wave per 16 rows
    k_gemm_wmma<<<blocks(gemmThreads), B, 0, stream>>>(
        hin, Bswz + (size_t)L * 2 * 4 * 32 * 16, bufT, agg, deg, n);
    k_scatter  <<<blocks((long long)E * 32), B, 0, stream>>>(agg, bufT, nrm, rowp, colp, E);
    k_bn_relu  <<<blocks((long long)n * (D / 4)), B, 0, stream>>>(agg, bs + L * D, gammas + L * D,
                                                                  betas + L * D, means + L * D,
                                                                  vars + L * D, n);
    hin = agg;
  }

  // ---- final linear head ----
  k_final<<<blocks((long long)n * 32), B, 0, stream>>>(out, hin, lin_w, lin_b, n);
}
